// GGUFLayer_63264868270569
// MI455X (gfx1250) — compile-verified, matrix operand order checked by reading, not run on hardware
//
#include <hip/hip_runtime.h>

typedef __attribute__((ext_vector_type(16))) __bf16        v16bf;
typedef __attribute__((ext_vector_type(8)))  float         v8f;
typedef __attribute__((ext_vector_type(4)))  unsigned int  v4u;
typedef __attribute__((ext_vector_type(8)))  int           v8i;
typedef __attribute__((ext_vector_type(4)))  int           v4i;

#define MT 128          // tokens per workgroup tile
#define NT 128          // output channels per workgroup tile
#define KT 64           // K step (two bf16 WMMA k-blocks)
#define LDST 72         // LDS row stride in bf16 elems (144B: 16B-aligned, conflict-free)

#if defined(__has_builtin)
#  if __has_builtin(__builtin_amdgcn_tensor_load_to_lds)
#    define HAVE_TDM 1
#  endif
#endif
#ifndef HAVE_TDM
#  define HAVE_TDM 0
#endif

#if defined(__has_builtin)
#  if __has_builtin(__builtin_amdgcn_s_wait_tensorcnt)
#    define WAIT_TENSOR(n) __builtin_amdgcn_s_wait_tensorcnt(n)
#  endif
#endif
#ifndef WAIT_TENSOR
#  define WAIT_TENSOR(n) asm volatile("s_wait_tensorcnt %0" ::"n"(n))
#endif

union FragBF16 {
    v16bf v;
    uint4 q[2];
};

__device__ __forceinline__ unsigned short f2bf(float f) {
    union { float f; unsigned int u; } c; c.f = f;
    unsigned int u = c.u;
    u += 0x7FFFu + ((u >> 16) & 1u);   // round-to-nearest-even
    return (unsigned short)(u >> 16);
}

__device__ __forceinline__ unsigned int pack2(unsigned short lo, unsigned short hi) {
    return (unsigned int)lo | ((unsigned int)hi << 16);
}

__device__ __forceinline__ unsigned lds_off(const unsigned short* p) {
    typedef __attribute__((address_space(3))) const unsigned short* lds_p;
    return (unsigned)(size_t)(lds_p)p;
}

// ---------------------------------------------------------------------------
// Pre-pass 1: x fp32 -> bf16 (row-major [T][I])
// ---------------------------------------------------------------------------
__global__ __launch_bounds__(256)
void cvt_x_bf16(const float* __restrict__ x, unsigned short* __restrict__ xb) {
    const size_t base = ((size_t)blockIdx.x * 256 + threadIdx.x) * 8;
    float4 a = *(const float4*)(x + base);
    float4 b = *(const float4*)(x + base + 4);
    uint4 o;
    o.x = pack2(f2bf(a.x), f2bf(a.y));
    o.y = pack2(f2bf(a.z), f2bf(a.w));
    o.z = pack2(f2bf(b.x), f2bf(b.y));
    o.w = pack2(f2bf(b.z), f2bf(b.w));
    *(uint4*)(xb + base) = o;
}

// ---------------------------------------------------------------------------
// Pre-pass 2: w Q8_0 dequant int32 -> bf16 (row-major [O][I]); scale idx = e/32
// ---------------------------------------------------------------------------
__global__ __launch_bounds__(256)
void dequant_w_bf16(const int* __restrict__ wq, const float* __restrict__ scales,
                    unsigned short* __restrict__ wb) {
    const size_t e = ((size_t)blockIdx.x * 256 + threadIdx.x) * 4;
    int4 q = *(const int4*)(wq + e);
    const float s = scales[e >> 5];
    uint2 o;
    o.x = pack2(f2bf((float)(q.x - 128) * s), f2bf((float)(q.y - 128) * s));
    o.y = pack2(f2bf((float)(q.z - 128) * s), f2bf((float)(q.w - 128) * s));
    *(uint2*)(wb + e) = o;
}

// ---------------------------------------------------------------------------
// TDM descriptor issue: DMA a 128-row x 64-col bf16 tile into LDS with
// hardware padding: 32 DWORDs (128B row) then +4 DWORDs -> 72-elem stride.
// ---------------------------------------------------------------------------
#if HAVE_TDM
__device__ __forceinline__ void tdm_load_tile(unsigned lds_byte,
                                              const unsigned short* gsrc,
                                              unsigned tensor_rows,
                                              unsigned tensor_cols) {
    const unsigned long long ga = (unsigned long long)(const void*)gsrc;
    const unsigned ga_lo = (unsigned)__builtin_amdgcn_readfirstlane((int)(unsigned)ga);
    const unsigned ga_hi = (unsigned)__builtin_amdgcn_readfirstlane((int)(unsigned)(ga >> 32));
    const unsigned ldsb  = (unsigned)__builtin_amdgcn_readfirstlane((int)lds_byte);

    // D# group 0: count=1 | lds_addr | global_addr[56:0] | type=2
    v4u g0 = { 1u, ldsb, ga_lo, (ga_hi & 0x01FFFFFFu) | 0x80000000u };

    // D# group 1
    const unsigned w0 = (1u << 16)      // data_size = 2 bytes
                      | (1u << 20)      // pad_enable
                      | (4u << 22)      // pad_interval: 32 DWORDs (128B)
                      | (3u << 25);     // pad_amount: 4 DWORDs (16B)
    const unsigned w1 = (tensor_cols & 0xFFFFu) << 16;                         // tensor_dim0 lo
    const unsigned w2 = (tensor_cols >> 16) | ((tensor_rows & 0xFFFFu) << 16); // dim0 hi | dim1 lo
    const unsigned w3 = (tensor_rows >> 16) | (64u << 16);                     // dim1 hi | tile_dim0=64
    const unsigned w4 = 128u;                                                  // tile_dim1=128, tile_dim2=0
    const unsigned w5 = tensor_cols;                                           // tensor_dim0_stride lo
    v8i g1  = { (int)w0, (int)w1, (int)w2, (int)w3, (int)w4, (int)w5, 0, 0 };
    v4i gz4 = { 0, 0, 0, 0 };
    v8i gz8 = { 0, 0, 0, 0, 0, 0, 0, 0 };
    // 6-arg variant (clang-23 / therock-10.0 headers)
    __builtin_amdgcn_tensor_load_to_lds(g0, g1, gz4, gz4, gz8, 0);
}
#endif

// ---------------------------------------------------------------------------
// Main GEMM: out[T,O] = xb[T,I](bf16) * wb[O,I](bf16)^T + bias, fp32 accum.
// 8 waves: 4x2 grid, each wave 32x64 via 2x4 accumulators; K-step 64.
// Double-buffered LDS tiles, TDM loads issued by wave 0 only.
// ---------------------------------------------------------------------------
__global__ __launch_bounds__(256)
void gemm_bf16_wmma(const unsigned short* __restrict__ xb,
                    const unsigned short* __restrict__ wb,
                    const float* __restrict__ bias,
                    float* __restrict__ out,
                    int T, int I, int O)
{
    __shared__ __align__(16) unsigned short xs[2][MT * LDST];
    __shared__ __align__(16) unsigned short wsh[2][NT * LDST];

    const int tid  = threadIdx.x;
    const int lane = tid & 31;
    const int wid  = tid >> 5;
    const int wm   = wid & 3;
    const int wn   = wid >> 2;
    const int half = (lane >> 4) & 1;
    const int l16  = lane & 15;

    const int wgN = blockIdx.x * NT;
    const int wgM = blockIdx.y * MT;
    const int nkb = I / KT;

    v8f acc[2][4];
#pragma unroll
    for (int mi = 0; mi < 2; ++mi)
#pragma unroll
        for (int nj = 0; nj < 4; ++nj)
            acc[mi][nj] = (v8f){0.f, 0.f, 0.f, 0.f, 0.f, 0.f, 0.f, 0.f};

#if HAVE_TDM
    unsigned xs_off[2], ws_off[2];
    xs_off[0] = lds_off(&xs[0][0]);  xs_off[1] = lds_off(&xs[1][0]);
    ws_off[0] = lds_off(&wsh[0][0]); ws_off[1] = lds_off(&wsh[1][0]);

    // wave-uniform issuer flag -> scalar branch (TDM ignores EXEC)
    const bool issuer = (__builtin_amdgcn_readfirstlane(wid) == 0);
    if (issuer) {
        tdm_load_tile(xs_off[0], xb + (size_t)wgM * I, (unsigned)T, (unsigned)I);
        tdm_load_tile(ws_off[0], wb + (size_t)wgN * I, (unsigned)O, (unsigned)I);
    }
#endif

    for (int kb = 0; kb < nkb; ++kb) {
        const int buf = kb & 1;

#if HAVE_TDM
        if (issuer) {
            if (kb + 1 < nkb) {
                const int nb = (kb + 1) & 1;
                const size_t koff = (size_t)(kb + 1) * KT;
                tdm_load_tile(xs_off[nb], xb + (size_t)wgM * I + koff, (unsigned)T, (unsigned)I);
                tdm_load_tile(ws_off[nb], wb + (size_t)wgN * I + koff, (unsigned)O, (unsigned)I);
                WAIT_TENSOR(2);   // the 2 just-issued may remain; tile kb complete
            } else {
                WAIT_TENSOR(0);
            }
        }
        __syncthreads();          // tile kb visible to all waves
#else
        __syncthreads();          // all waves done with previous use of this buffer
#pragma unroll
        for (int i = 0; i < 4; ++i) {
            const int c = tid + i * 256;
            const int r = c >> 3;
            const int ch = (c & 7) * 8;
            *(uint4*)&xs[buf][r * LDST + ch] =
                *(const uint4*)(xb + (size_t)(wgM + r) * I + (size_t)kb * KT + ch);
            *(uint4*)&wsh[buf][r * LDST + ch] =
                *(const uint4*)(wb + (size_t)(wgN + r) * I + (size_t)kb * KT + ch);
        }
        __syncthreads();
#endif

        const unsigned short* xt = xs[buf];
        const unsigned short* wt = wsh[buf];

#pragma unroll
        for (int ks = 0; ks < 2; ++ks) {
            FragBF16 a[2], b[4];
#pragma unroll
            for (int mi = 0; mi < 2; ++mi) {
                const int row = wm * 32 + mi * 16 + l16;
                const int e0  = row * LDST + ks * 32 + 8 * half;
                a[mi].q[0] = *(const uint4*)&xt[e0];
                a[mi].q[1] = *(const uint4*)&xt[e0 + 16];
            }
#pragma unroll
            for (int nj = 0; nj < 4; ++nj) {
                const int row = wn * 64 + nj * 16 + l16;
                const int e0  = row * LDST + ks * 32 + 8 * half;
                b[nj].q[0] = *(const uint4*)&wt[e0];
                b[nj].q[1] = *(const uint4*)&wt[e0 + 16];
            }
#pragma unroll
            for (int mi = 0; mi < 2; ++mi)
#pragma unroll
                for (int nj = 0; nj < 4; ++nj)
                    acc[mi][nj] = __builtin_amdgcn_wmma_f32_16x16x32_bf16(
                        false, a[mi].v, false, b[nj].v,
                        (short)0, acc[mi][nj], false, false);
        }

#if HAVE_TDM
        __syncthreads();   // everyone done reading buf before it is re-filled
#endif
    }

#pragma unroll
    for (int nj = 0; nj < 4; ++nj) {
        const int col = wgN + wn * 64 + nj * 16 + l16;
        const float bv = bias[col];
#pragma unroll
        for (int mi = 0; mi < 2; ++mi) {
            const int rbase = wgM + wm * 32 + mi * 16 + half * 8;
#pragma unroll
            for (int r = 0; r < 8; ++r)
                out[(size_t)(rbase + r) * O + col] = acc[mi][nj][r] + bv;
        }
    }
}

// ---------------------------------------------------------------------------
// Fallback: fused single-pass kernel (round-1 design), used if d_ws too small.
// ---------------------------------------------------------------------------
#define FLDST 40
__global__ __launch_bounds__(256)
void gguf_q8_linear_fused(const float* __restrict__ x,
                          const int*   __restrict__ wq,
                          const float* __restrict__ scales,
                          const float* __restrict__ bias,
                          float* __restrict__ out,
                          int T, int I, int O)
{
    __shared__ __align__(16) unsigned short xs[MT * FLDST];
    __shared__ __align__(16) unsigned short wsh[NT * FLDST];

    const int tid  = threadIdx.x;
    const int lane = tid & 31;
    const int wid  = tid >> 5;
    const int wm   = wid & 3;
    const int wn   = wid >> 2;
    const int half = (lane >> 4) & 1;
    const int l16  = lane & 15;

    const int wgN = blockIdx.x * NT;
    const int wgM = blockIdx.y * MT;
    const int nkb = I / 32;

    v8f acc[2][4];
#pragma unroll
    for (int mi = 0; mi < 2; ++mi)
#pragma unroll
        for (int nj = 0; nj < 4; ++nj)
            acc[mi][nj] = (v8f){0.f, 0.f, 0.f, 0.f, 0.f, 0.f, 0.f, 0.f};

    for (int kb = 0; kb < nkb; ++kb) {
        const int k0 = kb * 32;
        float4 xr[4]; int4 wr[4]; float sc[4]; int rw[4], cl[4];
#pragma unroll
        for (int i = 0; i < 4; ++i) {
            int c = tid + i * 256;
            int r = c >> 3, cc = (c & 7) * 4;
            rw[i] = r; cl[i] = cc;
            xr[i] = *(const float4*)(x  + (size_t)(wgM + r) * I + k0 + cc);
            wr[i] = *(const int4*)  (wq + (size_t)(wgN + r) * I + k0 + cc);
            sc[i] = scales[(size_t)(wgN + r) * nkb + kb];
        }
        __syncthreads();
#pragma unroll
        for (int i = 0; i < 4; ++i) {
            const int off = rw[i] * FLDST + cl[i];
            uint2 px;
            px.x = pack2(f2bf(xr[i].x), f2bf(xr[i].y));
            px.y = pack2(f2bf(xr[i].z), f2bf(xr[i].w));
            *(uint2*)&xs[off] = px;
            const float s = sc[i];
            uint2 pw;
            pw.x = pack2(f2bf((float)(wr[i].x - 128) * s), f2bf((float)(wr[i].y - 128) * s));
            pw.y = pack2(f2bf((float)(wr[i].z - 128) * s), f2bf((float)(wr[i].w - 128) * s));
            *(uint2*)&wsh[off] = pw;
        }
        __syncthreads();

        FragBF16 a[2], b[4];
#pragma unroll
        for (int mi = 0; mi < 2; ++mi) {
            const int row = wm * 32 + mi * 16 + l16;
            const int e0  = row * FLDST + 8 * half;
            a[mi].q[0] = *(const uint4*)&xs[e0];
            a[mi].q[1] = *(const uint4*)&xs[e0 + 16];
        }
#pragma unroll
        for (int nj = 0; nj < 4; ++nj) {
            const int row = wn * 64 + nj * 16 + l16;
            const int e0  = row * FLDST + 8 * half;
            b[nj].q[0] = *(const uint4*)&wsh[e0];
            b[nj].q[1] = *(const uint4*)&wsh[e0 + 16];
        }
#pragma unroll
        for (int mi = 0; mi < 2; ++mi)
#pragma unroll
            for (int nj = 0; nj < 4; ++nj)
                acc[mi][nj] = __builtin_amdgcn_wmma_f32_16x16x32_bf16(
                    false, a[mi].v, false, b[nj].v,
                    (short)0, acc[mi][nj], false, false);
    }

#pragma unroll
    for (int nj = 0; nj < 4; ++nj) {
        const int col = wgN + wn * 64 + nj * 16 + l16;
        const float bv = bias[col];
#pragma unroll
        for (int mi = 0; mi < 2; ++mi) {
            const int rbase = wgM + wm * 32 + mi * 16 + half * 8;
#pragma unroll
            for (int r = 0; r < 8; ++r)
                out[(size_t)(rbase + r) * O + col] = acc[mi][nj][r] + bv;
        }
    }
}

extern "C" void kernel_launch(void* const* d_in, const int* in_sizes, int n_in,
                              void* d_out, int out_size, void* d_ws, size_t ws_size,
                              hipStream_t stream) {
    const float* x      = (const float*)d_in[0];
    const int*   wq     = (const int*)  d_in[1];
    const float* scales = (const float*)d_in[2];
    const float* bias   = (const float*)d_in[3];
    float*       out    = (float*)d_out;

    const int O = in_sizes[3];            // 4096
    const int I = in_sizes[1] / O;        // 4096
    const int T = in_sizes[0] / I;        // 8192

    const size_t wbytes = (size_t)O * I * sizeof(unsigned short);
    const size_t xbytes = (size_t)T * I * sizeof(unsigned short);

    if (ws_size >= wbytes + xbytes) {
        unsigned short* wb = (unsigned short*)d_ws;
        unsigned short* xb = wb + (size_t)O * I;

        dequant_w_bf16<<<dim3((unsigned)((size_t)O * I / 1024)), dim3(256), 0, stream>>>(wq, scales, wb);
        cvt_x_bf16    <<<dim3((unsigned)((size_t)T * I / 2048)), dim3(256), 0, stream>>>(x, xb);

        dim3 grid(O / NT, T / MT), block(256);
        gemm_bf16_wmma<<<grid, block, 0, stream>>>(xb, wb, bias, out, T, I, O);
    } else {
        dim3 grid(O / NT, T / MT), block(256);
        gguf_q8_linear_fused<<<grid, block, 0, stream>>>(x, wq, scales, bias, out, T, I, O);
    }
}
